// TransformerBlock_43310450212962
// MI455X (gfx1250) — matrix operand order repaired and checked by reading, not compiled, not run on hardware
//
#include <hip/hip_runtime.h>
#include <hip/hip_bf16.h>
#include <math.h>

// ---------------------------------------------------------------------------
// Types for CDNA5 WMMA (wave32, 16x16x32 bf16 -> f32)
// ---------------------------------------------------------------------------
typedef __bf16 bf16_t;
typedef __attribute__((ext_vector_type(16))) __bf16 v16bf;
typedef __attribute__((ext_vector_type(8)))  __bf16 v8bf;
typedef __attribute__((ext_vector_type(8)))  float  v8f;
typedef __attribute__((ext_vector_type(4)))  unsigned int u32x4;
typedef __attribute__((ext_vector_type(8)))  int i32x8;
typedef __attribute__((ext_vector_type(4)))  int i32x4;

static_assert(sizeof(bf16_t) == 2, "bf16 must be 2 bytes");

#define DIMC     1024
#define HEADS    16
#define HEAD_DIM 64
#define HIDDEN   4096
#define BATCH    2
#define SEQ      2048
#define MTOT     (BATCH * SEQ)   // 4096 rows

static __device__ __forceinline__ v8f vzero8() {
    v8f z;
#pragma unroll
    for (int i = 0; i < 8; ++i) z[i] = 0.0f;
    return z;
}

static __device__ __forceinline__ v8f wmma_bf16(v16bf a, v16bf b, v8f c) {
    // (neg_a, A, neg_b, B, c_mod, C, reuse_a, reuse_b)
    return __builtin_amdgcn_wmma_f32_16x16x32_bf16(false, a, false, b, (short)0, c,
                                                   false, false);
}

// ---------------------------------------------------------------------------
// Tensor Data Mover: 2-D tile (tile1 rows x tile0 elems, bf16) global -> LDS.
// D# layout per CDNA5 ISA 8.3/8.4: group0 = {flags, lds_addr, global_addr, type},
// group1 = {mask/data_size, tensor dims, tile dims, dim0 stride}.
// Tensor dims set large: tiles here are never out of bounds (all sizes divide).
// This toolchain exposes the 6-arg builtin: (g0, g1, g2, g3, g4, cpol).
// ---------------------------------------------------------------------------
static __device__ __forceinline__ void tdm_load_2d(const void* gptr, void* lptr,
                                                   unsigned tile0, unsigned tile1,
                                                   unsigned stride_elems) {
    const unsigned lds = (unsigned)(size_t)lptr;   // low 32 bits = LDS offset
    const unsigned long long ga = (unsigned long long)(size_t)gptr;
    u32x4 g0;
    g0[0] = 1u;                                        // count=1 (valid), user mode
    g0[1] = lds;                                       // lds_addr
    g0[2] = (unsigned)(ga & 0xffffffffu);              // global_addr[31:0]
    g0[3] = (unsigned)((ga >> 32) & 0x01ffffffu) | (2u << 30);  // addr[56:32] | type=2
    const unsigned td0 = 0x40000000u, td1 = 0x40000000u;        // "infinite" tensor dims
    i32x8 g1;
    g1[0] = (int)(1u << 16);                           // workgroup_mask=0, data_size=1 (2B)
    g1[1] = (int)((td0 & 0xffffu) << 16);              // atomic_addr=0 | tensor_dim0 lo
    g1[2] = (int)((td0 >> 16) | ((td1 & 0xffffu) << 16));
    g1[3] = (int)((td1 >> 16) | (tile0 << 16));        // tensor_dim1 hi | tile_dim0
    g1[4] = (int)(tile1 & 0xffffu);                    // tile_dim1 ; tile_dim2=0
    const unsigned long long s0 = (unsigned long long)stride_elems;
    g1[5] = (int)(s0 & 0xffffffffu);                   // tensor_dim0_stride[31:0]
    g1[6] = (int)((s0 >> 32) & 0xffffu);               // stride[47:32] | dim1_stride lo=0
    g1[7] = 0;
    const i32x4 z4 = {0, 0, 0, 0};
    const i32x8 z8 = {0, 0, 0, 0, 0, 0, 0, 0};
    __builtin_amdgcn_tensor_load_to_lds(g0, g1, z4, z4, z8, 0);
}

static __device__ __forceinline__ void tdm_wait() {
    __builtin_amdgcn_s_wait_tensorcnt(0);
}

// A-fragment (16x32 bf16, MxK): lane -> M (lane&15).
// lanes 0-15 hold K = 0..7 and 16..23; lanes 16-31 hold K = 8..15 and 24..31.
static __device__ __forceinline__ v16bf load_frag_a(const bf16_t* base, int ld,
                                                    int row0, int k0, int lane) {
    const int m  = lane & 15;
    const int kb = (lane < 16) ? 0 : 8;
    const bf16_t* p = base + (size_t)(row0 + m) * ld + k0 + kb;
    union { v16bf v; v8bf h[2]; } u;
    u.h[0] = *(const v8bf*)(p);        // K = kb .. kb+7
    u.h[1] = *(const v8bf*)(p + 16);   // K = kb+16 .. kb+23
    return u.v;
}

// B-fragment (32x16 bf16, KxN): lane -> N (lane&15).
// lanes 0-15 hold K = 0..7 and 16..23; lanes 16-31 hold K = 8..15 and 24..31
// (same K order as the A-fragment). Source row-major [Nout, K].
static __device__ __forceinline__ v16bf load_frag_b(const bf16_t* base, int ld,
                                                    int row0, int k0, int lane) {
    const int n  = lane & 15;
    const int kb = (lane < 16) ? 0 : 8;
    const bf16_t* p = base + (size_t)(row0 + n) * ld + k0 + kb;
    union { v16bf v; v8bf h[2]; } u;
    u.h[0] = *(const v8bf*)(p);
    u.h[1] = *(const v8bf*)(p + 16);
    return u.v;
}

// ---------------------------------------------------------------------------
// fp32 -> bf16 conversion (weights)
// ---------------------------------------------------------------------------
__global__ void cvt_f32_bf16(const float* __restrict__ s, bf16_t* __restrict__ d, int n) {
    int i = blockIdx.x * blockDim.x + threadIdx.x;
    int stride = gridDim.x * blockDim.x;
    for (; i < n; i += stride) d[i] = (bf16_t)s[i];
}

// ---------------------------------------------------------------------------
// LayerNorm over DIMC, output bf16. One block (256 thr) per row.
// ---------------------------------------------------------------------------
__global__ __launch_bounds__(256)
void ln_to_bf16(const float* __restrict__ x, const float* __restrict__ w,
                const float* __restrict__ bvec, bf16_t* __restrict__ out) {
    __shared__ float red[256];
    const int row = blockIdx.x;
    const float* xr = x + (size_t)row * DIMC;
    float s = 0.f, s2 = 0.f;
    for (int i = threadIdx.x; i < DIMC; i += 256) {
        float v = xr[i];
        s += v; s2 += v * v;
    }
    red[threadIdx.x] = s;
    __syncthreads();
    for (int st = 128; st > 0; st >>= 1) {
        if (threadIdx.x < st) red[threadIdx.x] += red[threadIdx.x + st];
        __syncthreads();
    }
    const float mean = red[0] * (1.0f / DIMC);
    __syncthreads();
    red[threadIdx.x] = s2;
    __syncthreads();
    for (int st = 128; st > 0; st >>= 1) {
        if (threadIdx.x < st) red[threadIdx.x] += red[threadIdx.x + st];
        __syncthreads();
    }
    const float var  = red[0] * (1.0f / DIMC) - mean * mean;
    const float rstd = rsqrtf(var + 1e-5f);
    for (int i = threadIdx.x; i < DIMC; i += 256)
        out[(size_t)row * DIMC + i] = (bf16_t)((xr[i] - mean) * rstd * w[i] + bvec[i]);
}

// ---------------------------------------------------------------------------
// WMMA GEMM with TDM-staged LDS tiles:
//   C[M,Nout] = A[M,K](bf16) @ W[Nout,K](bf16)^T + bias, + epilogue.
// 256 thr = 8 waves; block tile 128x128; wave tile 32x64; K staged 32 wide,
// double-buffered: wave 0 drives two TENSOR_LOAD_TO_LDS per K-step.
// ---------------------------------------------------------------------------
enum { EPI_QKV = 0, EPI_RESID = 1, EPI_GELU = 2 };

template <int MODE>
__global__ __launch_bounds__(256)
void gemm_wmma(const bf16_t* __restrict__ A, const bf16_t* __restrict__ W,
               const float* __restrict__ bias, int M, int Nout, int K,
               float* __restrict__ out_f32, bf16_t* __restrict__ out_bf16,
               const float* __restrict__ resid, const float* __restrict__ gamma,
               bf16_t* __restrict__ qbuf, bf16_t* __restrict__ kbuf,
               bf16_t* __restrict__ vtbuf) {
    __shared__ __align__(16) bf16_t sA[2][128][32];
    __shared__ __align__(16) bf16_t sB[2][128][32];

    const int lane = threadIdx.x & 31;
    const int wid  = threadIdx.x >> 5;
    const int rowBase = blockIdx.y * 128;
    const int colBase = blockIdx.x * 128;
    const int m0 = (wid & 3) * 32;   // within block tile
    const int n0 = (wid >> 2) * 64;

    const int nk = K >> 5;
    if (wid == 0) {
        tdm_load_2d(A + (size_t)rowBase * K, &sA[0][0][0], 32, 128, K);
        tdm_load_2d(W + (size_t)colBase * K, &sB[0][0][0], 32, 128, K);
    }

    v8f acc[2][4];
#pragma unroll
    for (int i = 0; i < 2; ++i)
#pragma unroll
        for (int j = 0; j < 4; ++j) acc[i][j] = vzero8();

    if (wid == 0) tdm_wait();
    __syncthreads();

    for (int kt = 0; kt < nk; ++kt) {
        const int cur = kt & 1;
        if (kt + 1 < nk && wid == 0) {
            const int nxt = cur ^ 1;
            tdm_load_2d(A + (size_t)rowBase * K + (size_t)(kt + 1) * 32,
                        &sA[nxt][0][0], 32, 128, K);
            tdm_load_2d(W + (size_t)colBase * K + (size_t)(kt + 1) * 32,
                        &sB[nxt][0][0], 32, 128, K);
        }
        const v16bf a0 = load_frag_a(&sA[cur][0][0], 32, m0,      0, lane);
        const v16bf a1 = load_frag_a(&sA[cur][0][0], 32, m0 + 16, 0, lane);
#pragma unroll
        for (int nt = 0; nt < 4; ++nt) {
            const v16bf b = load_frag_b(&sB[cur][0][0], 32, n0 + nt * 16, 0, lane);
            acc[0][nt] = wmma_bf16(a0, b, acc[0][nt]);
            acc[1][nt] = wmma_bf16(a1, b, acc[1][nt]);
        }
        if (kt + 1 < nk) {
            if (wid == 0) tdm_wait();
            __syncthreads();
        }
    }

    // C/D layout: lane -> N = lane&15; VGPR r -> M = r + 8*(lane>=16)
    const int half = lane >> 4;
    const int nl   = lane & 15;
#pragma unroll
    for (int mt = 0; mt < 2; ++mt) {
#pragma unroll
        for (int nt = 0; nt < 4; ++nt) {
            const int n = colBase + n0 + nt * 16 + nl;
            const float bn = bias[n];
#pragma unroll
            for (int r = 0; r < 8; ++r) {
                const int m = rowBase + m0 + mt * 16 + half * 8 + r;
                float v = acc[mt][nt][r] + bn;
                if (MODE == EPI_QKV) {
                    const int which = n >> 10;       // 0=q 1=k 2=v
                    const int c     = n & 1023;
                    const int head  = c >> 6;
                    const int d     = c & 63;
                    const int bb    = m >> 11;
                    const int nr    = m & 2047;
                    if (which == 0) {
                        qbuf[((size_t)(bb * HEADS + head) * SEQ + nr) * HEAD_DIM + d] =
                            (bf16_t)(v * 0.125f);    // fold 1/sqrt(64) into Q
                    } else if (which == 1) {
                        kbuf[((size_t)(bb * HEADS + head) * SEQ + nr) * HEAD_DIM + d] =
                            (bf16_t)v;
                    } else {
                        vtbuf[((size_t)(bb * HEADS + head) * HEAD_DIM + d) * SEQ + nr] =
                            (bf16_t)v;               // V stored transposed [B,H,D,N]
                    }
                } else if (MODE == EPI_RESID) {
                    const size_t idx = (size_t)m * Nout + n;
                    out_f32[idx] = resid[idx] + gamma[n] * v;
                } else { // EPI_GELU (exact, erf-based)
                    const float g = 0.5f * v * (1.0f + erff(v * 0.70710678118654752f));
                    out_bf16[(size_t)m * Nout + n] = (bf16_t)g;
                }
            }
        }
    }
}

// ---------------------------------------------------------------------------
// Flash attention, TDM-staged: grid (SEQ/128, B*H); 8 waves x 16 query rows.
// 64 keys per step; K/V^T tiles staged into LDS once per block (not per wave),
// double-buffered via TDM. Row l kept as per-lane partials (one final shuffle
// reduction); row max needs 4 shuffles per row per 64 keys.
// ---------------------------------------------------------------------------
__global__ __launch_bounds__(256)
void attn_kernel(const bf16_t* __restrict__ qbuf, const bf16_t* __restrict__ kbuf,
                 const bf16_t* __restrict__ vtbuf, bf16_t* __restrict__ obuf) {
    __shared__ __align__(16) bf16_t sK[2][64][64];   // [keyLocal][d]
    __shared__ __align__(16) bf16_t sV[2][64][64];   // [d][keyLocal]
    __shared__ __align__(16) bf16_t pbuf[8][16][64]; // per-wave P tile

    const int lane = threadIdx.x & 31;
    const int wid  = threadIdx.x >> 5;
    const int half = lane >> 4;
    const int nl   = lane & 15;

    const int bh   = blockIdx.y;
    const int b    = bh >> 4;
    const int head = bh & 15;
    const int q0   = blockIdx.x * 128 + wid * 16;

    const bf16_t* qbh = qbuf  + (size_t)bh * SEQ * HEAD_DIM;
    const bf16_t* kbh = kbuf  + (size_t)bh * SEQ * HEAD_DIM;
    const bf16_t* vbh = vtbuf + (size_t)bh * HEAD_DIM * SEQ;

    const v16bf qf0 = load_frag_a(qbh, HEAD_DIM, q0, 0,  lane);
    const v16bf qf1 = load_frag_a(qbh, HEAD_DIM, q0, 32, lane);

    const int NIT = SEQ / 64;  // 32 key-tile steps
    if (wid == 0) {
        tdm_load_2d(kbh, &sK[0][0][0], 64, 64, HEAD_DIM);
        tdm_load_2d(vbh, &sV[0][0][0], 64, 64, SEQ);
    }

    v8f oacc[4];
#pragma unroll
    for (int nt = 0; nt < 4; ++nt) oacc[nt] = vzero8();
    float mrun[8], lrun[8];
#pragma unroll
    for (int r = 0; r < 8; ++r) { mrun[r] = -1e30f; lrun[r] = 0.0f; }

    if (wid == 0) tdm_wait();
    __syncthreads();

    for (int it = 0; it < NIT; ++it) {
        const int j0  = it * 64;
        const int cur = it & 1;
        if (it + 1 < NIT && wid == 0) {
            const int nxt = cur ^ 1;
            tdm_load_2d(kbh + (size_t)(j0 + 64) * HEAD_DIM, &sK[nxt][0][0], 64, 64, HEAD_DIM);
            tdm_load_2d(vbh + (size_t)(j0 + 64),            &sV[nxt][0][0], 64, 64, SEQ);
        }

        // scores: 16 queries x 64 keys (4 subtiles); scale folded into Q
        v8f s[4];
#pragma unroll
        for (int c = 0; c < 4; ++c) {
            s[c] = vzero8();
            const v16bf k0f = load_frag_b(&sK[cur][0][0], 64, c * 16, 0,  lane);
            const v16bf k1f = load_frag_b(&sK[cur][0][0], 64, c * 16, 32, lane);
            s[c] = wmma_bf16(qf0, k0f, s[c]);
            s[c] = wmma_bf16(qf1, k1f, s[c]);
        }

        // online softmax update (row lives in one 16-lane half at VGPR r)
#pragma unroll
        for (int r = 0; r < 8; ++r) {
            float rowmax = fmaxf(fmaxf(s[0][r], s[1][r]), fmaxf(s[2][r], s[3][r]));
#pragma unroll
            for (int off = 8; off >= 1; off >>= 1)
                rowmax = fmaxf(rowmax, __shfl_xor(rowmax, off, 32));
            const float mnew = fmaxf(mrun[r], rowmax);
            const float corr = __expf(mrun[r] - mnew);
            mrun[r] = mnew;
            float psum = 0.0f;
            const int mrow = half * 8 + r;
#pragma unroll
            for (int c = 0; c < 4; ++c) {
                const float p = __expf(s[c][r] - mnew);
                psum += p;
                pbuf[wid][mrow][c * 16 + nl] = (bf16_t)p;
            }
            lrun[r] = lrun[r] * corr + psum;   // per-lane partial row sum
#pragma unroll
            for (int nt = 0; nt < 4; ++nt) oacc[nt][r] *= corr;
        }

        // P (16x64) as two A fragments from LDS (same-wave RAW: DS in order)
        const v16bf pf0 = load_frag_a(&pbuf[wid][0][0], 64, 0, 0,  lane);
        const v16bf pf1 = load_frag_a(&pbuf[wid][0][0], 64, 0, 32, lane);

        // O += P @ V (V^T rows are contiguous in sV)
#pragma unroll
        for (int nt = 0; nt < 4; ++nt) {
            const v16bf vf0 = load_frag_b(&sV[cur][0][0], 64, nt * 16, 0,  lane);
            const v16bf vf1 = load_frag_b(&sV[cur][0][0], 64, nt * 16, 32, lane);
            oacc[nt] = wmma_bf16(pf0, vf0, oacc[nt]);
            oacc[nt] = wmma_bf16(pf1, vf1, oacc[nt]);
        }

        if (it + 1 < NIT) {
            if (wid == 0) tdm_wait();
            __syncthreads();
        }
    }

    // final row-sum reduction (once) + normalize + store O rows of [MTOT,DIMC]
#pragma unroll
    for (int r = 0; r < 8; ++r) {
        float rs = lrun[r];
#pragma unroll
        for (int off = 8; off >= 1; off >>= 1)
            rs += __shfl_xor(rs, off, 32);
        const float inv = 1.0f / rs;
        const int row = b * SEQ + q0 + half * 8 + r;
#pragma unroll
        for (int nt = 0; nt < 4; ++nt) {
            const int col = head * HEAD_DIM + nt * 16 + nl;
            obuf[(size_t)row * DIMC + col] = (bf16_t)(oacc[nt][r] * inv);
        }
    }
}

// ---------------------------------------------------------------------------
// Host launcher
// ---------------------------------------------------------------------------
extern "C" void kernel_launch(void* const* d_in, const int* in_sizes, int n_in,
                              void* d_out, int out_size, void* d_ws, size_t ws_size,
                              hipStream_t stream) {
    (void)in_sizes; (void)n_in; (void)out_size; (void)ws_size;

    const float* x      = (const float*)d_in[0];
    const float* ln1_w  = (const float*)d_in[1];
    const float* ln1_b  = (const float*)d_in[2];
    const float* qkv_w  = (const float*)d_in[3];
    const float* qkv_b  = (const float*)d_in[4];
    const float* proj_w = (const float*)d_in[5];
    const float* proj_b = (const float*)d_in[6];
    const float* ls1_g  = (const float*)d_in[7];
    const float* ln2_w  = (const float*)d_in[8];
    const float* ln2_b  = (const float*)d_in[9];
    const float* fc1_w  = (const float*)d_in[10];
    const float* fc1_b  = (const float*)d_in[11];
    const float* fc2_w  = (const float*)d_in[12];
    const float* fc2_b  = (const float*)d_in[13];
    const float* ls2_g  = (const float*)d_in[14];

    char* ws = (char*)d_ws;
    size_t off = 0;
    auto alloc = [&](size_t bytes) -> void* {
        void* p = ws + off;
        off += (bytes + 255) & ~(size_t)255;
        return p;
    };

    bf16_t* qkv_w_bf  = (bf16_t*)alloc((size_t)3 * DIMC * DIMC * 2);
    bf16_t* proj_w_bf = (bf16_t*)alloc((size_t)DIMC * DIMC * 2);
    bf16_t* fc1_w_bf  = (bf16_t*)alloc((size_t)HIDDEN * DIMC * 2);
    bf16_t* fc2_w_bf  = (bf16_t*)alloc((size_t)DIMC * HIDDEN * 2);
    bf16_t* h_bf      = (bf16_t*)alloc((size_t)MTOT * DIMC * 2);
    bf16_t* q_bf      = (bf16_t*)alloc((size_t)MTOT * DIMC * 2);
    bf16_t* k_bf      = (bf16_t*)alloc((size_t)MTOT * DIMC * 2);
    bf16_t* vt_bf     = (bf16_t*)alloc((size_t)MTOT * DIMC * 2);
    bf16_t* o_bf      = (bf16_t*)alloc((size_t)MTOT * DIMC * 2);
    float*  x1        = (float*)alloc((size_t)MTOT * DIMC * 4);
    bf16_t* h2_bf     = (bf16_t*)alloc((size_t)MTOT * DIMC * 2);
    bf16_t* g_bf      = (bf16_t*)alloc((size_t)MTOT * HIDDEN * 2);

    // weights -> bf16
    cvt_f32_bf16<<<1024, 256, 0, stream>>>(qkv_w,  qkv_w_bf,  3 * DIMC * DIMC);
    cvt_f32_bf16<<<512,  256, 0, stream>>>(proj_w, proj_w_bf, DIMC * DIMC);
    cvt_f32_bf16<<<2048, 256, 0, stream>>>(fc1_w,  fc1_w_bf,  HIDDEN * DIMC);
    cvt_f32_bf16<<<2048, 256, 0, stream>>>(fc2_w,  fc2_w_bf,  DIMC * HIDDEN);

    // LN1
    ln_to_bf16<<<MTOT, 256, 0, stream>>>(x, ln1_w, ln1_b, h_bf);

    // QKV: [4096,3072] = h @ qkv_w^T ; scatter into q/k/v^T
    gemm_wmma<EPI_QKV><<<dim3(3 * DIMC / 128, MTOT / 128), 256, 0, stream>>>(
        h_bf, qkv_w_bf, qkv_b, MTOT, 3 * DIMC, DIMC,
        nullptr, nullptr, nullptr, nullptr, q_bf, k_bf, vt_bf);

    // attention
    attn_kernel<<<dim3(SEQ / 128, BATCH * HEADS), 256, 0, stream>>>(q_bf, k_bf, vt_bf, o_bf);

    // proj + residual with ls1_gamma -> x1 (f32)
    gemm_wmma<EPI_RESID><<<dim3(DIMC / 128, MTOT / 128), 256, 0, stream>>>(
        o_bf, proj_w_bf, proj_b, MTOT, DIMC, DIMC,
        x1, nullptr, x, ls1_g, nullptr, nullptr, nullptr);

    // LN2
    ln_to_bf16<<<MTOT, 256, 0, stream>>>(x1, ln2_w, ln2_b, h2_bf);

    // FC1 + exact GELU -> bf16
    gemm_wmma<EPI_GELU><<<dim3(HIDDEN / 128, MTOT / 128), 256, 0, stream>>>(
        h2_bf, fc1_w_bf, fc1_b, MTOT, HIDDEN, DIMC,
        nullptr, g_bf, nullptr, nullptr, nullptr, nullptr, nullptr);

    // FC2 + residual with ls2_gamma -> d_out (f32)
    gemm_wmma<EPI_RESID><<<dim3(DIMC / 128, MTOT / 128), 256, 0, stream>>>(
        g_bf, fc2_w_bf, fc2_b, MTOT, DIMC, HIDDEN,
        (float*)d_out, nullptr, x1, ls2_g, nullptr, nullptr, nullptr);
}
